// EmbeddingPredictor_75471165325381
// MI455X (gfx1250) — matrix operand order repaired and checked by reading, not compiled, not run on hardware
//
#include <hip/hip_runtime.h>
#include <hip/hip_bf16.h>

// Problem constants (match reference)
#define VV 8000
#define EE 1024
#define HH 8
#define CC 3
#define BB 16
#define TT 2048
#define BT (BB * TT)
#define EPS 1e-5f

typedef __attribute__((ext_vector_type(16))) __bf16 v16bf;
typedef __attribute__((ext_vector_type(8)))  __bf16 v8bf;
typedef __attribute__((ext_vector_type(8)))  float  v8f;

union FragAB {
    v16bf v;
    v8bf  h[2];
};

__device__ __forceinline__ unsigned short f32_to_bf16_rne(float f) {
    union { float f; unsigned u; } x;
    x.f = f;
    unsigned r = x.u + 0x7fffu + ((x.u >> 16) & 1u);
    return (unsigned short)(r >> 16);
}

// ---------------------------------------------------------------------------
// Kernel 1: P[c][e] = sum_h pos_w[h,e,c]  (head-folded position weights)
//           Wb      = bf16(ffn_w)          (weights for WMMA, 2MB -> L2 resident)
// grid: EE blocks x 256 threads; block n converts weight row n; blocks 0..2
// additionally produce P row c=n.
// ---------------------------------------------------------------------------
__global__ void prep_kernel(const float* __restrict__ pos_w,
                            const float* __restrict__ ffn_w,
                            unsigned short* __restrict__ Wb,
                            float* __restrict__ P) {
    const int n   = blockIdx.x;
    const int tid = threadIdx.x;

    const float* wr = ffn_w + (size_t)n * EE;
    for (int i = tid; i < EE; i += 256)
        Wb[(size_t)n * EE + i] = f32_to_bf16_rne(wr[i]);

    if (n < CC) {
        for (int e = tid; e < EE; e += 256) {
            float s = 0.f;
#pragma unroll
            for (int h = 0; h < HH; ++h)
                s += pos_w[(size_t)h * EE * CC + (size_t)e * CC + n];
            P[n * EE + e] = s;
        }
    }
}

// ---------------------------------------------------------------------------
// Kernel 2: fused gather + window mixing.
// One block per (b,t) token. Each thread owns 4 embedding columns.
//   s_c  = dot(win_c, P_c)            (block reduction, 3 scalars)
//   mix  = (s0*win0 + s1*win1 + s2*win2) / 24   -> bf16 into ws (X matrix)
// ---------------------------------------------------------------------------
__global__ void mix_kernel(const int*   __restrict__ tokens,
                           const float* __restrict__ table,
                           const float* __restrict__ P,
                           unsigned short* __restrict__ Xb) {
    const int row = blockIdx.x;          // b*T + t
    const int t   = row % TT;
    const int tid = threadIdx.x;
    const int e0  = tid * 4;

    const int* tokrow = tokens + (row - t);   // tokens + b*T

    float4 v[3];
    float  p[3];

#pragma unroll
    for (int c = 0; c < CC; ++c) {
        const int tt = t + c - (CC - 1);
        if (tt >= 0) {
            const int tok = tokrow[tt];
            v[c] = *(const float4*)(table + (size_t)tok * EE + e0);
        } else {
            v[c] = make_float4(0.f, 0.f, 0.f, 0.f);
        }
        const float4 pc = *(const float4*)(P + c * EE + e0);
        p[c] = v[c].x * pc.x + v[c].y * pc.y + v[c].z * pc.z + v[c].w * pc.w;
    }

    // wave32 reduction then cross-wave via LDS
#pragma unroll
    for (int c = 0; c < CC; ++c)
#pragma unroll
        for (int mask = 16; mask; mask >>= 1)
            p[c] += __shfl_xor(p[c], mask, 32);

    __shared__ float red[CC][8];
    __shared__ float sbc[CC];
    const int wave = tid >> 5, lane = tid & 31;
    if (lane == 0) {
#pragma unroll
        for (int c = 0; c < CC; ++c) red[c][wave] = p[c];
    }
    __syncthreads();
    if (tid < CC) {
        float s = 0.f;
#pragma unroll
        for (int w = 0; w < 8; ++w) s += red[tid][w];
        sbc[tid] = s;
    }
    __syncthreads();

    const float inv = 1.0f / (float)(HH * CC);
    const float s0 = sbc[0] * inv, s1 = sbc[1] * inv, s2 = sbc[2] * inv;

    float4 m;
    m.x = s0 * v[0].x + s1 * v[1].x + s2 * v[2].x;
    m.y = s0 * v[0].y + s1 * v[1].y + s2 * v[2].y;
    m.z = s0 * v[0].z + s1 * v[1].z + s2 * v[2].z;
    m.w = s0 * v[0].w + s1 * v[1].w + s2 * v[2].w;

    ushort4 o;
    o.x = f32_to_bf16_rne(m.x);
    o.y = f32_to_bf16_rne(m.y);
    o.z = f32_to_bf16_rne(m.z);
    o.w = f32_to_bf16_rne(m.w);
    *(ushort4*)(Xb + (size_t)row * EE + e0) = o;
}

// ---------------------------------------------------------------------------
// Kernel 3: bf16 WMMA GEMM  out = X @ W^T + b, fused LayerNorm + swish.
// Block: 16 output rows x 1024 cols, 256 threads = 8 waves, wave w owns
// columns [w*128, w*128+128). 32 K-steps x 8 tiles -> 256 v_wmma per wave.
//
// Fragment layouts per ISA 7.12.2 (16-bit, wave32):
//   A 16x32: lane holds row m=lane%16; elems = K {8h..8h+7} ++ {16+8h..23+8h}
//            -> two 16B loads from the row-major X row.
//   B 32x16: lane holds col n=lane%16; elems = K {8h..8h+7} ++ {16+8h..23+8h}, same as A
//            -> two 16B loads from the row-major W row (W[n][k], since
//               out[m,n] = sum_k X[m,k]*W[n,k]).
//   C/D 16x16 f32: vgpr j, lane l -> M = j + 8*(l>>4), N = l%16.
// ---------------------------------------------------------------------------
__global__ void __launch_bounds__(256)
gemm_ln_swish_kernel(const unsigned short* __restrict__ Xb,
                     const unsigned short* __restrict__ Wb,
                     const float* __restrict__ bias,
                     const float* __restrict__ ln_g,
                     const float* __restrict__ ln_b,
                     float* __restrict__ out) {
    __shared__ float ldsT[16][EE];   // 64 KB output staging tile

    const int m0   = blockIdx.x * 16;
    const int tid  = threadIdx.x;
    const int wave = tid >> 5;
    const int lane = tid & 31;
    const int half = lane >> 4;
    const int l16  = lane & 15;
    const int nwv  = wave * 128;

    v8f acc[8] = {};

    const __bf16* X = (const __bf16*)Xb;
    const __bf16* W = (const __bf16*)Wb;
    const __bf16* xrow = X + (size_t)(m0 + l16) * EE;

    for (int k = 0; k < EE; k += 32) {
        __builtin_prefetch(xrow + k + 32, 0, 1);   // global_prefetch next A chunk
        FragAB a;
        a.h[0] = *(const v8bf*)(xrow + k + 8 * half);
        a.h[1] = *(const v8bf*)(xrow + k + 16 + 8 * half);
#pragma unroll
        for (int t = 0; t < 8; ++t) {
            const __bf16* wrow = W + (size_t)(nwv + t * 16 + l16) * EE;
            FragAB b;
            b.h[0] = *(const v8bf*)(wrow + k + 8 * half);
            b.h[1] = *(const v8bf*)(wrow + k + 16 + 8 * half);
            acc[t] = __builtin_amdgcn_wmma_f32_16x16x32_bf16(
                false, a.v, false, b.v, (short)0, acc[t], false, false);
        }
    }

    // bias add + dump accumulators into LDS tile
#pragma unroll
    for (int t = 0; t < 8; ++t) {
        const int n  = nwv + t * 16 + l16;
        const float bn = bias[n];
#pragma unroll
        for (int j = 0; j < 8; ++j) {
            const int m = j + 8 * half;
            ldsT[m][n] = acc[t][j] + bn;
        }
    }
    __syncthreads();

    // Per-row LayerNorm stats + normalize + swish, in place.
    // 256 threads -> 16 groups of 16 lanes; group handles row = tid>>4.
    {
        const int r  = tid >> 4;
        const int c0 = tid & 15;
        float s = 0.f, q = 0.f;
        for (int n = c0; n < EE; n += 16) {
            const float v = ldsT[r][n];
            s += v;
            q += v * v;
        }
#pragma unroll
        for (int mask = 8; mask; mask >>= 1) {
            s += __shfl_xor(s, mask, 32);
            q += __shfl_xor(q, mask, 32);
        }
        const float mu  = s * (1.0f / EE);
        const float var = q * (1.0f / EE) - mu * mu;
        const float rs  = rsqrtf(var + EPS);
        for (int n = c0; n < EE; n += 16) {
            const float y = (ldsT[r][n] - mu) * rs * ln_g[n] + ln_b[n];
            ldsT[r][n] = y / (1.0f + __expf(-y));   // swish
        }
    }
    __syncthreads();

    // Coalesced float4 writeout: thread owns 4 fixed columns across 16 rows.
    const int c = tid * 4;
#pragma unroll
    for (int r = 0; r < 16; ++r) {
        *(float4*)(out + (size_t)(m0 + r) * EE + c) = *(const float4*)&ldsT[r][c];
    }
}

// ---------------------------------------------------------------------------
// Launcher
// ---------------------------------------------------------------------------
extern "C" void kernel_launch(void* const* d_in, const int* in_sizes, int n_in,
                              void* d_out, int out_size, void* d_ws, size_t ws_size,
                              hipStream_t stream) {
    const int*   tokens = (const int*)  d_in[0];
    const float* table  = (const float*)d_in[1];
    const float* pos_w  = (const float*)d_in[2];
    const float* ffn_w  = (const float*)d_in[3];
    const float* ffn_b  = (const float*)d_in[4];
    const float* ln_g   = (const float*)d_in[5];
    const float* ln_b   = (const float*)d_in[6];
    float* out = (float*)d_out;

    // workspace layout: [X bf16 64MB][W bf16 2MB][P f32 12KB]
    unsigned short* Xb = (unsigned short*)d_ws;
    unsigned short* Wb = Xb + (size_t)BT * EE;
    float*          P  = (float*)(Wb + (size_t)EE * EE);

    prep_kernel<<<EE, 256, 0, stream>>>(pos_w, ffn_w, Wb, P);
    mix_kernel<<<BT, 256, 0, stream>>>(tokens, table, P, Xb);
    gemm_ln_swish_kernel<<<BT / 16, 256, 0, stream>>>(Xb, Wb, ffn_b, ln_g, ln_b, out);
}
